// PlanarPFL_67834713473355
// MI455X (gfx1250) — matrix operand order repaired and checked by reading, not compiled, not run on hardware
//
#include <hip/hip_runtime.h>
#include <hip/hip_bf16.h>

#define BATCH 16384
#define DIM   768
#define NFLOW 4
#define NSAMP 8

typedef __attribute__((ext_vector_type(16))) __bf16 v16bf;
typedef __attribute__((ext_vector_type(8)))  float  v8f;

union FragU { unsigned int u[8]; v16bf v; };

__device__ __forceinline__ unsigned short f32_bf16(float f) {
  unsigned int u = __float_as_uint(f);
  u += 0x7FFFu + ((u >> 16) & 1u);       // round-to-nearest-even
  return (unsigned short)(u >> 16);
}

__device__ __forceinline__ float softplusf(float x) {
  return x > 20.f ? x : log1pf(__expf(x));
}

__device__ __forceinline__ float waveSum(float v) {
#pragma unroll
  for (int off = 16; off > 0; off >>= 1) v += __shfl_xor(v, off, 32);
  return v;
}

// ---------------- conversion kernels ----------------

__global__ __launch_bounds__(256)
void cvt_bf16_kernel(const float* __restrict__ in, unsigned short* __restrict__ out, int n) {
  int i = blockIdx.x * 256 + threadIdx.x;
  if (i < n) out[i] = f32_bf16(in[i]);
}

// out[n*rows + k] = in[k*cols + n]   (weight [K,N] -> bf16 [N,K])
__global__ __launch_bounds__(256)
void cvtT_bf16_kernel(const float* __restrict__ in, unsigned short* __restrict__ out,
                      int rows, int cols) {
  int i = blockIdx.x * 256 + threadIdx.x;
  if (i < rows * cols) {
    int n = i / rows, k = i - n * rows;
    out[i] = f32_bf16(in[(size_t)k * cols + n]);
  }
}

// b = x @ amor_b_w + amor_b_b   -> [B,4]
__global__ __launch_bounds__(256)
void amorb_kernel(const float* __restrict__ x, const float* __restrict__ w,
                  const float* __restrict__ bb, float* __restrict__ out) {
  int idx = blockIdx.x * 256 + threadIdx.x;   // < B*4
  int row = idx >> 2, k = idx & 3;
  float acc = bb[k];
  const float* xr = x + (size_t)row * DIM;
  for (int d = 0; d < DIM; ++d) acc += xr[d] * w[d * 4 + k];
  out[idx] = acc;
}

// ---------------- WMMA GEMM: C = act(A_bf16 @ Wt_bf16^T + bias) ----------------
// A:[M,Kd] bf16 row-major, Wt:[N,Kd] bf16 (pre-transposed). Block = 8 waves,
// tiles 128(M) x 64(N); each wave: one A-frag x 4 B-frags per 32-k step.
// Explicit ping-pong double buffering: fragment loads for step kb are issued
// BEFORE the WMMAs consuming step kb-32, so each WMMA group only needs a
// partial loadcnt wait and the load clause overlaps the matrix pipeline.
// Requires Kd % 64 == 0 (Kd = 768 here).
// zmode=1 scatters columns into the [8][M][DIM] z-region (slabBase + col/DIM).
__global__ __launch_bounds__(256)
void wmma_gemm_kernel(const unsigned short* __restrict__ A,
                      const unsigned short* __restrict__ Wt,
                      const float* __restrict__ bias,
                      float* __restrict__ outF,
                      unsigned short* __restrict__ outB,
                      int M, int N, int Kd, int act, int zmode, int slabBase) {
  const int tid    = threadIdx.x;
  const int wid    = tid >> 5;
  const int lane   = tid & 31;
  const int lanehi = lane >> 4;
  const int l16    = lane & 15;
  const int m0     = blockIdx.y * 128 + wid * 16;
  const int n0     = blockIdx.x * 64;

  v8f acc[4] = {};

  int koA[8], koB[8];
#pragma unroll
  for (int v = 0; v < 8; ++v) {
    koA[v] = ((v >= 4) ? 16 : 0) + (lanehi ? 8 : 0) + 2 * (v & 3); // 16-bit A 16x32 layout
    koB[v] = ((v >= 4) ? 16 : 0) + (lanehi ? 8 : 0) + 2 * (v & 3); // 16-bit B 32x16 layout
  }

  const unsigned short* Arow = A + (size_t)(m0 + l16) * Kd;
  const unsigned short* Brow[4];
#pragma unroll
  for (int s = 0; s < 4; ++s)
    Brow[s] = Wt + (size_t)(n0 + s * 16 + l16) * Kd;

  FragU a0, a1, b0[4], b1[4];

  auto loadSet = [&](FragU& aa, FragU* bb, int kb) {
#pragma unroll
    for (int v = 0; v < 8; ++v)
      aa.u[v] = *(const unsigned int*)(Arow + kb + koA[v]);
#pragma unroll
    for (int s = 0; s < 4; ++s)
#pragma unroll
      for (int v = 0; v < 8; ++v)
        bb[s].u[v] = *(const unsigned int*)(Brow[s] + kb + koB[v]);
  };
  auto mmSet = [&](FragU& aa, FragU* bb) {
#pragma unroll
    for (int s = 0; s < 4; ++s)
      acc[s] = __builtin_amdgcn_wmma_f32_16x16x32_bf16(
          false, aa.v, false, bb[s].v, (short)0, acc[s], false, false);
  };

  const int nk = Kd >> 5;          // 32-wide k-steps; even
  loadSet(a0, b0, 0);
  int kb = 32;
#pragma unroll 1
  for (int it = 0; it < (nk >> 1) - 1; ++it) {
    loadSet(a1, b1, kb);           // prefetch odd step
    mmSet(a0, b0);                 // consume even step (in flight since last iter)
    loadSet(a0, b0, kb + 32);      // prefetch next even step
    mmSet(a1, b1);                 // consume odd step
    kb += 64;
  }
  loadSet(a1, b1, kb);             // last odd step
  mmSet(a0, b0);
  mmSet(a1, b1);

#pragma unroll
  for (int s = 0; s < 4; ++s) {
#pragma unroll
    for (int r = 0; r < 8; ++r) {
      int row = m0 + r + 8 * lanehi;      // C/D layout: VGPR r, lane-half selects M
      int col = n0 + s * 16 + l16;
      float v = acc[s][r] + bias[col];
      if (act == 1) v = softplusf(v);
      if (zmode) {
        int slab = slabBase + col / DIM;
        outF[(size_t)slab * M * DIM + (size_t)row * DIM + (col % DIM)] = v;
      } else {
        if (outF) outF[(size_t)row * N + col] = v;
        if (outB) outB[(size_t)row * N + col] = f32_bf16(v);
      }
    }
  }
}

// ---------------- fused planar flows (all 4 flows, z in LDS) ----------------
// One workgroup per batch row. zreg = d_out z-region, [8][B][D]:
// on entry slabs 0-3 hold u[b][k][:], slabs 4-7 hold w[b][k][:] (from zmode GEMMs);
// on exit slabs hold z_k[l][b][:]. Each block touches only its own row in every slab.
__global__ __launch_bounds__(256)
void flow_kernel(const float* __restrict__ eps,
                 const float* __restrict__ mu,
                 const float* __restrict__ var,   // already softplus'd
                 const float* __restrict__ bvec,  // [B,4]
                 float* __restrict__ zreg,
                 unsigned short* __restrict__ zk7b) {
  const int bidx = blockIdx.x;
  const int tid  = threadIdx.x;
  const int wid  = tid >> 5;
  const int lane = tid & 31;

  __shared__ float zsh[NSAMP * DIM];
  __shared__ float wk[DIM];
  __shared__ float uh[DIM];
  __shared__ float red[2][8];
  __shared__ float tl[NSAMP];
  __shared__ float scsh;

  const size_t slabSz = (size_t)BATCH * DIM;
  const float* murow  = mu  + (size_t)bidx * DIM;
  const float* varrow = var + (size_t)bidx * DIM;

  // reparameterized samples z0 = eps * sqrt(var) + mu
  for (int idx = tid; idx < NSAMP * DIM; idx += 256) {
    int l = idx / DIM, d = idx - l * DIM;
    zsh[idx] = eps[((size_t)l * BATCH + bidx) * DIM + d] * sqrtf(varrow[d]) + murow[d];
  }
  __syncthreads();

  for (int k = 0; k < NFLOW; ++k) {
    for (int d = tid; d < DIM; d += 256) {
      uh[d] = zreg[(size_t)k * slabSz + (size_t)bidx * DIM + d];
      wk[d] = zreg[(size_t)(NFLOW + k) * slabSz + (size_t)bidx * DIM + d];
    }
    __syncthreads();

    float puw = 0.f, pww = 0.f;
    for (int d = tid; d < DIM; d += 256) {
      float wv = wk[d];
      puw += wv * uh[d];
      pww += wv * wv;
    }
    puw = waveSum(puw);
    pww = waveSum(pww);
    if (lane == 0) { red[0][wid] = puw; red[1][wid] = pww; }
    __syncthreads();
    if (tid == 0) {
      float uw = 0.f, ww = 0.f;
      for (int i = 0; i < 8; ++i) { uw += red[0][i]; ww += red[1][i]; }
      float m = -1.f + softplusf(uw);
      scsh = (m - uw) / ww;
    }
    __syncthreads();

    float sc = scsh;
    for (int d = tid; d < DIM; d += 256) uh[d] += sc * wk[d];  // u_hat

    // wzb: wave w reduces sample l == w (8 waves <-> L=8)
    float p = 0.f;
    for (int d = lane; d < DIM; d += 32) p += zsh[wid * DIM + d] * wk[d];
    p = waveSum(p);
    if (lane == 0) tl[wid] = tanhf(p + bvec[(size_t)bidx * NFLOW + k]);
    __syncthreads();   // also orders uh writes before the update below

    for (int idx = tid; idx < NSAMP * DIM; idx += 256) {
      int l = idx / DIM, d = idx - l * DIM;
      zsh[idx] += uh[d] * tl[l];
    }
    __syncthreads();
  }

  for (int idx = tid; idx < NSAMP * DIM; idx += 256) {
    int l = idx / DIM, d = idx - l * DIM;
    zreg[(size_t)l * slabSz + (size_t)bidx * DIM + d] = zsh[idx];
  }
  for (int d = tid; d < DIM; d += 256)
    zk7b[(size_t)bidx * DIM + d] = f32_bf16(zsh[7 * DIM + d]);
}

// ---------------- host launcher ----------------

extern "C" void kernel_launch(void* const* d_in, const int* in_sizes, int n_in,
                              void* d_out, int out_size, void* d_ws, size_t ws_size,
                              hipStream_t stream) {
  (void)in_sizes; (void)n_in; (void)out_size; (void)ws_size;

  const float* x         = (const float*)d_in[0];
  const float* eps       = (const float*)d_in[1];
  const float* enc_mu_w  = (const float*)d_in[2];
  const float* enc_mu_b  = (const float*)d_in[3];
  const float* enc_var_w = (const float*)d_in[4];
  const float* enc_var_b = (const float*)d_in[5];
  const float* amor_u_w  = (const float*)d_in[6];
  const float* amor_u_b  = (const float*)d_in[7];
  const float* amor_w_w  = (const float*)d_in[8];
  const float* amor_w_b  = (const float*)d_in[9];
  const float* amor_b_w  = (const float*)d_in[10];
  const float* amor_b_b  = (const float*)d_in[11];
  const float* dec_w     = (const float*)d_in[12];
  const float* dec_b     = (const float*)d_in[13];
  const float* pmu_w     = (const float*)d_in[14];
  const float* pmu_b     = (const float*)d_in[15];

  float* xmean = (float*)d_out;                         // [B,D]; doubles as mu staging
  float* zreg  = (float*)d_out + (size_t)BATCH * DIM;   // [8][B][D]

  char* ws = (char*)d_ws;
  size_t off = 0;
  auto alloc = [&](size_t bytes) -> char* {
    char* p = ws + off;
    off = (off + bytes + 255) & ~(size_t)255;
    return p;
  };

  unsigned short* xb     = (unsigned short*)alloc((size_t)BATCH * DIM * 2);
  unsigned short* wt_mu  = (unsigned short*)alloc((size_t)DIM * DIM * 2);
  unsigned short* wt_var = (unsigned short*)alloc((size_t)DIM * DIM * 2);
  unsigned short* wt_u   = (unsigned short*)alloc((size_t)DIM * NFLOW * DIM * 2);
  unsigned short* wt_w   = (unsigned short*)alloc((size_t)DIM * NFLOW * DIM * 2);
  unsigned short* wt_dec = (unsigned short*)alloc((size_t)DIM * DIM * 2);
  unsigned short* wt_pmu = (unsigned short*)alloc((size_t)DIM * DIM * 2);
  float*          varb   = (float*)alloc((size_t)BATCH * DIM * 4);
  float*          bvec   = (float*)alloc((size_t)BATCH * NFLOW * 4);
  unsigned short* zk7b   = (unsigned short*)alloc((size_t)BATCH * DIM * 2);
  unsigned short* hb     = (unsigned short*)alloc((size_t)BATCH * DIM * 2);

  const int nBD = BATCH * DIM;
  cvt_bf16_kernel<<<(nBD + 255) / 256, 256, 0, stream>>>(x, xb, nBD);

  const int nDD = DIM * DIM, nDK = DIM * NFLOW * DIM;
  cvtT_bf16_kernel<<<(nDD + 255) / 256, 256, 0, stream>>>(enc_mu_w,  wt_mu,  DIM, DIM);
  cvtT_bf16_kernel<<<(nDD + 255) / 256, 256, 0, stream>>>(enc_var_w, wt_var, DIM, DIM);
  cvtT_bf16_kernel<<<(nDK + 255) / 256, 256, 0, stream>>>(amor_u_w,  wt_u,   DIM, NFLOW * DIM);
  cvtT_bf16_kernel<<<(nDK + 255) / 256, 256, 0, stream>>>(amor_w_w,  wt_w,   DIM, NFLOW * DIM);
  cvtT_bf16_kernel<<<(nDD + 255) / 256, 256, 0, stream>>>(dec_w,     wt_dec, DIM, DIM);
  cvtT_bf16_kernel<<<(nDD + 255) / 256, 256, 0, stream>>>(pmu_w,     wt_pmu, DIM, DIM);

  amorb_kernel<<<(BATCH * NFLOW) / 256, 256, 0, stream>>>(x, amor_b_w, amor_b_b, bvec);

  dim3 blk(256);
  dim3 g768(DIM / 64, BATCH / 128);
  dim3 g3072((NFLOW * DIM) / 64, BATCH / 128);

  // mu -> x_mean region (overwritten later); var -> softplus'd workspace
  wmma_gemm_kernel<<<g768, blk, 0, stream>>>(xb, wt_mu,  enc_mu_b,  xmean, nullptr,
                                             BATCH, DIM, DIM, 0, 0, 0);
  wmma_gemm_kernel<<<g768, blk, 0, stream>>>(xb, wt_var, enc_var_b, varb,  nullptr,
                                             BATCH, DIM, DIM, 1, 0, 0);
  // u -> z-region slabs 0-3, w -> slabs 4-7
  wmma_gemm_kernel<<<g3072, blk, 0, stream>>>(xb, wt_u, amor_u_b, zreg, nullptr,
                                              BATCH, NFLOW * DIM, DIM, 0, 1, 0);
  wmma_gemm_kernel<<<g3072, blk, 0, stream>>>(xb, wt_w, amor_w_b, zreg, nullptr,
                                              BATCH, NFLOW * DIM, DIM, 0, 1, NFLOW);

  flow_kernel<<<BATCH, 256, 0, stream>>>(eps, xmean, varb, bvec, zreg, zk7b);

  // h = zk[-1] @ dec_w + dec_b (bf16 only), then x_mean = h @ pmu_w + pmu_b
  wmma_gemm_kernel<<<g768, blk, 0, stream>>>(zk7b, wt_dec, dec_b, nullptr, hb,
                                             BATCH, DIM, DIM, 0, 0, 0);
  wmma_gemm_kernel<<<g768, blk, 0, stream>>>(hb, wt_pmu, pmu_b, xmean, nullptr,
                                             BATCH, DIM, DIM, 0, 0, 0);
}
